// Block_69088843924067
// MI455X (gfx1250) — matrix operand order repaired and checked by reading, not compiled, not run on hardware
//
#include <hip/hip_runtime.h>

// ---------------------------------------------------------------------------
// Types for CDNA5 WMMA (wave32): A/B = 16 bf16 (8 VGPRs), C/D = 8 f32.
// ---------------------------------------------------------------------------
typedef __attribute__((ext_vector_type(16))) __bf16 v16bf;
typedef __attribute__((ext_vector_type(8)))  float  v8f;

__device__ __forceinline__ v8f wmma_bf16(v16bf a, v16bf b, v8f c) {
  // D = A(16x32) * B(32x16) + C, f32 accumulate
  return __builtin_amdgcn_wmma_f32_16x16x32_bf16(
      /*neg_a=*/false, a, /*neg_b=*/false, b,
      /*c_mod=*/(short)0, c, /*reuse_a=*/false, /*reuse_b=*/false);
}

// A-matrix 16x32 (MxK), row-major source, leading dim ld (elements).
// lane m=L&15; g=L>>4; VGPR i: K=16g+2i (contiguous order, matches B loaders)
__device__ __forceinline__ v16bf load_a_frag(const __bf16* src, int ld) {
  const int lane = threadIdx.x & 31;
  const int m = lane & 15, g = lane >> 4;
  const __bf16* row = src + m * ld;
  v16bf r;
#pragma unroll
  for (int i = 0; i < 8; ++i) {
    int k0 = 16 * g + 2 * i;
    r[2 * i]     = row[k0];
    r[2 * i + 1] = row[k0 + 1];
  }
  return r;
}

// B-matrix 32x16 (KxN), row-major source (B[k][n] = src[k*ld+n]).  (LDS use)
__device__ __forceinline__ v16bf load_b_frag(const __bf16* src, int ld) {
  const int lane = threadIdx.x & 31;
  const int n = lane & 15, g = lane >> 4;
  v16bf r;
#pragma unroll
  for (int j = 0; j < 8; ++j) {
    int k = 16 * g + 2 * j;
    r[2 * j]     = src[(size_t)k * ld + n];
    r[2 * j + 1] = src[(size_t)(k + 1) * ld + n];
  }
  return r;
}

// B-matrix 32x16 where logical B[k][n] = src[n*ld + k] (transposed source).
// Each lane reads 16 contiguous bf16 -> compiler fuses into b128 loads.
__device__ __forceinline__ v16bf load_bT_frag(const __bf16* src, int ld) {
  const int lane = threadIdx.x & 31;
  const int n = lane & 15, g = lane >> 4;
  const __bf16* row = src + (size_t)n * ld + 16 * g;
  v16bf r;
#pragma unroll
  for (int j = 0; j < 8; ++j) {
    r[2 * j]     = row[2 * j];
    r[2 * j + 1] = row[2 * j + 1];
  }
  return r;
}

// ---------------------------------------------------------------------------
// Generic bf16 GEMM:  C[M,N] = act(A[M,K] @ B[K,N] + bias + res)
// Block: 256 threads (8 waves), tile 128x128. Wave tile 32x64 (2x4 WMMA frags).
// Double-buffered LDS staging via CDNA5 async global->LDS DMA (ASYNCcnt),
// plus L2 prefetch hints 2 tiles ahead.
// ---------------------------------------------------------------------------
template <bool RELU, bool BIAS, bool RES, bool OUTF, bool OUTB>
__global__ __launch_bounds__(256) void gemm_bf16(
    const __bf16* __restrict__ A, const __bf16* __restrict__ B,
    const float* __restrict__ bias, const float* __restrict__ res,
    float* __restrict__ outF, __bf16* __restrict__ outB,
    int M, int N, int K) {
  constexpr int LDA = 40;    // 32 + 8 pad (keeps 16B alignment)
  constexpr int LDB = 136;   // 128 + 8 pad
  __shared__ __bf16 As[2][128 * LDA];
  __shared__ __bf16 Bs[2][32 * LDB];

  const int t = threadIdx.x;
  const int wave = t >> 5, lane = t & 31;
  const int wm = wave & 3, wn = wave >> 2;
  const int m0 = blockIdx.y * 128, n0 = blockIdx.x * 128;

  v8f c[2][4] = {};

  // Async DMA: 16B chunks straight into LDS, no VGPR round-trip.
  // LDS operand = low 32 bits of the flat shared pointer (aperture keeps the
  // workgroup-relative offset in addr[31:0]).
  auto stage = [&](int buf, int k0) {
#pragma unroll
    for (int i = 0; i < 2; ++i) {        // A tile 128x32 : 512 vec8
      int v = t + i * 256;
      int row = v >> 2, col = (v & 3) * 8;
      unsigned lds = (unsigned)(uintptr_t)&As[buf][row * LDA + col];
      const __bf16* gp = &A[(size_t)(m0 + row) * K + k0 + col];
      asm volatile("global_load_async_to_lds_b128 %0, %1, off"
                   :: "v"(lds), "v"(gp) : "memory");
    }
#pragma unroll
    for (int i = 0; i < 2; ++i) {        // B tile 32x128 : 512 vec8
      int v = t + i * 256;
      int row = v >> 4, col = (v & 15) * 8;
      unsigned lds = (unsigned)(uintptr_t)&Bs[buf][row * LDB + col];
      const __bf16* gp = &B[(size_t)(k0 + row) * N + n0 + col];
      asm volatile("global_load_async_to_lds_b128 %0, %1, off"
                   :: "v"(lds), "v"(gp) : "memory");
    }
  };

  const int nk = K >> 5;
  stage(0, 0);
  asm volatile("s_wait_asynccnt 0" ::: "memory");
  __syncthreads();

  int buf = 0;
  for (int kt = 0; kt < nk; ++kt) {
    if (kt + 1 < nk) stage(buf ^ 1, (kt + 1) << 5);
    if (kt + 2 < nk) {  // L2 prefetch hints (global_prefetch_b8)
      int row = t >> 2, col = (t & 3) * 8;
      __builtin_prefetch(&A[(size_t)(m0 + row) * K + ((kt + 2) << 5) + col], 0, 1);
      int rb = t >> 4, cb = (t & 15) * 8;
      __builtin_prefetch(&B[(size_t)(((kt + 2) << 5) + rb) * N + n0 + cb], 0, 1);
    }
    const __bf16* aBase = &As[buf][(wm * 32) * LDA];
    v16bf a0 = load_a_frag(aBase, LDA);
    v16bf a1 = load_a_frag(aBase + 16 * LDA, LDA);
    const __bf16* bBase = &Bs[buf][wn * 64];
#pragma unroll
    for (int j = 0; j < 4; ++j) {
      v16bf bj = load_b_frag(bBase + j * 16, LDB);
      c[0][j] = wmma_bf16(a0, bj, c[0][j]);
      c[1][j] = wmma_bf16(a1, bj, c[1][j]);
    }
    asm volatile("s_wait_asynccnt 0" ::: "memory");
    __syncthreads();
    buf ^= 1;
  }

  // Epilogue (fused bias / residual / relu, dual-precision store)
  const int nl = lane & 15, mo = (lane >> 4) * 8;
#pragma unroll
  for (int i = 0; i < 2; ++i) {
#pragma unroll
    for (int j = 0; j < 4; ++j) {
      int gm = m0 + wm * 32 + i * 16 + mo;
      int gn = n0 + wn * 64 + j * 16 + nl;
      float bv = BIAS ? bias[gn] : 0.0f;
#pragma unroll
      for (int r = 0; r < 8; ++r) {
        size_t idx = (size_t)(gm + r) * N + gn;
        float v = c[i][j][r] + bv;
        if (RES)  v += res[idx];
        if (RELU) v = fmaxf(v, 0.0f);
        if (OUTF) outF[idx] = v;
        if (OUTB) outB[idx] = (__bf16)v;
      }
    }
  }
}

// ---------------------------------------------------------------------------
// LDS-tiled transpose: V [B*T, 512] (b,t,n) -> Vt [B*512, T] (b,n,t)
// Coalesced on both sides; one-shot 8MB pass (~1.5us at HBM speed) that turns
// every V-fragment load in the flash inner loop into contiguous b128 traffic.
// ---------------------------------------------------------------------------
__global__ __launch_bounds__(256) void transpose_bh(
    const __bf16* __restrict__ src, __bf16* __restrict__ dst, int T) {
  __shared__ __bf16 tile[32][33];
  const int b = blockIdx.z;
  const int n0 = blockIdx.y * 32;
  const int t0 = blockIdx.x * 32;
  const int x = threadIdx.x, y = threadIdx.y;   // 32 x 8
#pragma unroll
  for (int i = 0; i < 32; i += 8)
    tile[y + i][x] = src[(size_t)(b * T + t0 + y + i) * 512 + n0 + x];
  __syncthreads();
#pragma unroll
  for (int i = 0; i < 32; i += 8)
    dst[(size_t)(b * 512 + n0 + y + i) * T + t0 + x] = tile[x][y + i];
}

// ---------------------------------------------------------------------------
// Flash attention (causal, online softmax). Q/K stored [B*T, H*64] bf16,
// V stored transposed [B*H, 64, T] bf16. Block = 8 waves; each wave owns a
// 16-row Q tile, streams 32-key tiles. All K/V fragment loads are contiguous.
// ---------------------------------------------------------------------------
__global__ __launch_bounds__(256) void flash_attn(
    const __bf16* __restrict__ Q, const __bf16* __restrict__ Kt,
    const __bf16* __restrict__ Vt, __bf16* __restrict__ O,
    int T, float scale) {
  __shared__ __bf16 Pst[8][16 * 32];   // per-wave P staging tile

  const int wave = threadIdx.x >> 5;
  const int lane = threadIdx.x & 31;
  const int g = lane >> 4;             // C-frag lane group (rows 0-7 vs 8-15)
  const int nl = lane & 15;            // C-frag column within tile
  const int bh = blockIdx.y;
  const int b = bh >> 3;
  const int h = bh & 7;
  const int q0 = blockIdx.x * 128 + wave * 16;
  const int ld = 512;

  const __bf16* Qp = Q + ((size_t)b * T + q0) * ld + h * 64;
  v16bf aq0 = load_a_frag(Qp, ld);        // e = 0..31
  v16bf aq1 = load_a_frag(Qp + 32, ld);   // e = 32..63
#pragma unroll
  for (int i = 0; i < 16; ++i) {          // fold softmax scale into Q
    aq0[i] = (__bf16)((float)aq0[i] * scale);
    aq1[i] = (__bf16)((float)aq1[i] * scale);
  }

  v8f o[4] = {};
  float rmax[8], rsum[8];
#pragma unroll
  for (int r = 0; r < 8; ++r) { rmax[r] = -3.0e38f; rsum[r] = 0.0f; }

  __bf16* pst = &Pst[wave][0];
  const __bf16* Vtp = Vt + (size_t)bh * 64 * T;   // [64, T] for this (b,h)

  for (int j0 = 0; j0 < q0 + 16; j0 += 32) {
    const __bf16* Kp = Kt + ((size_t)b * T + j0) * ld + h * 64;

    // S = Q @ K^T  (16x32 tile, 4 WMMAs)
    v8f s[2] = {};
#pragma unroll
    for (int nb = 0; nb < 2; ++nb) {
      v16bf bk0 = load_bT_frag(Kp + (size_t)nb * 16 * ld, ld);
      v16bf bk1 = load_bT_frag(Kp + (size_t)nb * 16 * ld + 32, ld);
      s[nb] = wmma_bf16(aq0, bk0, s[nb]);
      s[nb] = wmma_bf16(aq1, bk1, s[nb]);
    }

    // causal mask: only the diagonal tile needs it (uniform branch per wave)
    if (j0 + 31 > q0) {
#pragma unroll
      for (int nb = 0; nb < 2; ++nb)
#pragma unroll
        for (int r = 0; r < 8; ++r) {
          int row = q0 + g * 8 + r;
          int col = j0 + nb * 16 + nl;
          if (col > row) s[nb][r] = -3.0e38f;
        }
    }

    // online softmax: new row max, rescale factor
    float al[8];
#pragma unroll
    for (int r = 0; r < 8; ++r) {
      float mx = fmaxf(s[0][r], s[1][r]);
      mx = fmaxf(mx, __shfl_xor(mx, 1, 32));
      mx = fmaxf(mx, __shfl_xor(mx, 2, 32));
      mx = fmaxf(mx, __shfl_xor(mx, 4, 32));
      mx = fmaxf(mx, __shfl_xor(mx, 8, 32));
      float nm = fmaxf(rmax[r], mx);
      al[r] = __expf(rmax[r] - nm);
      rmax[r] = nm;
    }
    // p = exp(s - m), row sums
#pragma unroll
    for (int nb = 0; nb < 2; ++nb)
#pragma unroll
      for (int r = 0; r < 8; ++r)
        s[nb][r] = __expf(s[nb][r] - rmax[r]);
#pragma unroll
    for (int r = 0; r < 8; ++r) {
      float sm = s[0][r] + s[1][r];
      sm += __shfl_xor(sm, 1, 32);
      sm += __shfl_xor(sm, 2, 32);
      sm += __shfl_xor(sm, 4, 32);
      sm += __shfl_xor(sm, 8, 32);
      rsum[r] = rsum[r] * al[r] + sm;
    }
    // rescale accumulated O
#pragma unroll
    for (int j = 0; j < 4; ++j)
#pragma unroll
      for (int r = 0; r < 8; ++r) o[j][r] *= al[r];

    // restage P (C-frag layout) -> LDS -> A-frag layout (same-wave, in-order DS)
#pragma unroll
    for (int nb = 0; nb < 2; ++nb)
#pragma unroll
      for (int r = 0; r < 8; ++r)
        pst[(g * 8 + r) * 32 + nb * 16 + nl] = (__bf16)s[nb][r];
    asm volatile("s_wait_dscnt 0" ::: "memory");
    v16bf ap = load_a_frag(pst, 32);

    // O += P @ V : B[k][e] = Vt[e][j0+k] -> contiguous per-lane loads
#pragma unroll
    for (int j = 0; j < 4; ++j) {
      v16bf bv = load_bT_frag(Vtp + (size_t)(j * 16) * T + j0, T);
      o[j] = wmma_bf16(ap, bv, o[j]);
    }
  }

  // normalize and store to concat layout [B*T, H*64] (bf16 -> proj GEMM input)
  __bf16* Op = O + ((size_t)b * T + q0) * 512 + h * 64;
#pragma unroll
  for (int r = 0; r < 8; ++r) {
    float inv = 1.0f / rsum[r];
#pragma unroll
    for (int j = 0; j < 4; ++j)
      Op[(size_t)(g * 8 + r) * 512 + j * 16 + nl] = (__bf16)(o[j][r] * inv);
  }
}

// ---------------------------------------------------------------------------
// fp32 -> bf16 converters / weight packers
// ---------------------------------------------------------------------------
__global__ void f32_to_bf16_v4(const float* __restrict__ s,
                               __bf16* __restrict__ d, int n) {
  int i = (blockIdx.x * blockDim.x + threadIdx.x) * 4;
  if (i < n) {
    float4 v = *(const float4*)(s + i);
    d[i] = (__bf16)v.x; d[i + 1] = (__bf16)v.y;
    d[i + 2] = (__bf16)v.z; d[i + 3] = (__bf16)v.w;
  }
}

// Wq [H=8, D=512, HS=64]  ->  [D=512, H*HS=512] bf16 (concat-head GEMM form)
__global__ void pack_qkv_w(const float* __restrict__ src,
                           __bf16* __restrict__ dst) {
  int i = blockIdx.x * blockDim.x + threadIdx.x;
  if (i < 512 * 512) {
    int d = i >> 9, n = i & 511;
    int h = n >> 6, e = n & 63;
    dst[i] = (__bf16)src[((size_t)h * 512 + d) * 64 + e];
  }
}

// ---------------------------------------------------------------------------
// Host launcher
// ---------------------------------------------------------------------------
extern "C" void kernel_launch(void* const* d_in, const int* in_sizes, int n_in,
                              void* d_out, int out_size, void* d_ws, size_t ws_size,
                              hipStream_t stream) {
  constexpr int B = 2, T = 4096, D = 512, H = 8;
  constexpr int BT = B * T;           // 8192
  constexpr int DFF = 4 * D;          // 2048

  const float* x     = (const float*)d_in[0];
  const float* Wq    = (const float*)d_in[1];
  const float* Wk    = (const float*)d_in[2];
  const float* Wv    = (const float*)d_in[3];
  const float* Wproj = (const float*)d_in[4];
  const float* bproj = (const float*)d_in[5];
  const float* W1    = (const float*)d_in[6];
  const float* b1    = (const float*)d_in[7];
  const float* W2    = (const float*)d_in[8];
  const float* b2    = (const float*)d_in[9];
  float* out = (float*)d_out;

  char* ws = (char*)d_ws;
  size_t off = 0;
  auto alloc = [&](size_t bytes) -> void* {
    off = (off + 255) & ~(size_t)255;
    void* p = ws + off;
    off += bytes;
    return p;
  };

  __bf16* Xbf  = (__bf16*)alloc((size_t)BT * D * 2);
  __bf16* Wqb  = (__bf16*)alloc((size_t)D * D * 2);
  __bf16* Wkb  = (__bf16*)alloc((size_t)D * D * 2);
  __bf16* Wvb  = (__bf16*)alloc((size_t)D * D * 2);
  __bf16* Wpb  = (__bf16*)alloc((size_t)D * D * 2);
  __bf16* W1b  = (__bf16*)alloc((size_t)D * DFF * 2);
  __bf16* W2b  = (__bf16*)alloc((size_t)DFF * D * 2);
  __bf16* Qb   = (__bf16*)alloc((size_t)BT * D * 2);
  __bf16* Kb   = (__bf16*)alloc((size_t)BT * D * 2);
  __bf16* Vb   = (__bf16*)alloc((size_t)BT * D * 2);
  __bf16* Vtb  = (__bf16*)alloc((size_t)BT * D * 2);   // V transposed [B*H,64,T]
  __bf16* Ab   = (__bf16*)alloc((size_t)BT * D * 2);
  float*  X1   = (float*) alloc((size_t)BT * D * 4);
  __bf16* X1b  = (__bf16*)alloc((size_t)BT * D * 2);
  __bf16* H1b  = (__bf16*)alloc((size_t)BT * DFF * 2);

  // --- convert / pack operands to bf16 ---
  f32_to_bf16_v4<<<(BT * D / 4 + 255) / 256, 256, 0, stream>>>(x, Xbf, BT * D);
  pack_qkv_w<<<(D * D) / 256, 256, 0, stream>>>(Wq, Wqb);
  pack_qkv_w<<<(D * D) / 256, 256, 0, stream>>>(Wk, Wkb);
  pack_qkv_w<<<(D * D) / 256, 256, 0, stream>>>(Wv, Wvb);
  f32_to_bf16_v4<<<(D * D / 4 + 255) / 256, 256, 0, stream>>>(Wproj, Wpb, D * D);
  f32_to_bf16_v4<<<(D * DFF / 4 + 255) / 256, 256, 0, stream>>>(W1, W1b, D * DFF);
  f32_to_bf16_v4<<<(DFF * D / 4 + 255) / 256, 256, 0, stream>>>(W2, W2b, DFF * D);

  // --- QKV projections: [8192,512] @ [512,512] -> bf16 ---
  dim3 g512(D / 128, BT / 128);
  gemm_bf16<false, false, false, false, true><<<g512, 256, 0, stream>>>(
      Xbf, Wqb, nullptr, nullptr, nullptr, Qb, BT, D, D);
  gemm_bf16<false, false, false, false, true><<<g512, 256, 0, stream>>>(
      Xbf, Wkb, nullptr, nullptr, nullptr, Kb, BT, D, D);
  gemm_bf16<false, false, false, false, true><<<g512, 256, 0, stream>>>(
      Xbf, Wvb, nullptr, nullptr, nullptr, Vb, BT, D, D);

  // --- transpose V for contiguous flash B-fragments ---
  transpose_bh<<<dim3(T / 32, D / 32, B), dim3(32, 8), 0, stream>>>(Vb, Vtb, T);

  // --- causal flash attention ---
  flash_attn<<<dim3(T / 128, B * H), 256, 0, stream>>>(
      Qb, Kb, Vtb, Ab, T, 0.125f /* HS^-0.5 */);

  // --- output projection + bias + residual: X1 = x + Attn@Wproj + bproj ---
  gemm_bf16<false, true, true, true, true><<<g512, 256, 0, stream>>>(
      Ab, Wpb, bproj, x, X1, X1b, BT, D, D);

  // --- FFN1: H1 = relu(X1 @ W1 + b1) ---
  gemm_bf16<true, true, false, false, true><<<dim3(DFF / 128, BT / 128), 256, 0, stream>>>(
      X1b, W1b, b1, nullptr, nullptr, H1b, BT, DFF, D);

  // --- FFN2 + residual: out = X1 + H1 @ W2 + b2 ---
  gemm_bf16<false, true, true, true, false><<<g512, 256, 0, stream>>>(
      H1b, W2b, b2, X1, out, nullptr, BT, D, DFF);
}
